// Attention_86371792322618
// MI455X (gfx1250) — matrix operand order repaired and checked by reading, not compiled, not run on hardware
//
#include <hip/hip_runtime.h>
#include <hip/hip_bf16.h>
#include <math.h>

// Shapes from the reference
#define HH   8
#define CH   32
#define CQ   256
#define SS   128
#define NQ   256
#define NKK  256

typedef __attribute__((ext_vector_type(16))) __bf16 v16bf;
typedef __attribute__((ext_vector_type(8)))  __bf16 v8bf;
typedef __attribute__((ext_vector_type(4)))  __bf16 v4bf;
typedef __attribute__((ext_vector_type(8)))  float  v8f;
typedef __attribute__((ext_vector_type(4)))  unsigned int v4u;
typedef __attribute__((ext_vector_type(4)))  int    v4i;
typedef __attribute__((ext_vector_type(8)))  int    v8i;

__device__ __forceinline__ v8f vzero8() {
    v8f z = {0.f, 0.f, 0.f, 0.f, 0.f, 0.f, 0.f, 0.f};
    return z;
}

__device__ __forceinline__ v8f wmma_bf16(v16bf a, v16bf b, v8f c) {
    // D = A(16x32 bf16) * B(32x16 bf16) + C(16x16 f32)
    return __builtin_amdgcn_wmma_f32_16x16x32_bf16(
        false, a, false, b, (short)0, c, false, false);
}

// A fragment (16x32, bf16): per lane two contiguous 16B runs
__device__ __forceinline__ v16bf load_a_frag(const __bf16* rowbase, int lh) {
    v8bf lo = *(const v8bf*)(rowbase + lh * 16);
    v8bf hi = *(const v8bf*)(rowbase + lh * 16 + 8);
    v16bf a;
    #pragma unroll
    for (int j = 0; j < 8; ++j) { a[j] = lo[j]; a[8 + j] = hi[j]; }
    return a;
}
// B fragment (32x16, bf16): per lane one contiguous 32B run
__device__ __forceinline__ v16bf load_b_frag(const __bf16* p) {
    return *(const v16bf*)p;
}

// ---------------------------------------------------------------------------
// Tensor Data Mover: 1D tile (n 2-byte elements) global -> LDS.
// Guarded: __has_builtin for availability, clang major for 5 vs 6 arg arity.
// ---------------------------------------------------------------------------
#if defined(__has_builtin)
#if __has_builtin(__builtin_amdgcn_tensor_load_to_lds) && \
    __has_builtin(__builtin_amdgcn_s_wait_tensorcnt)
#define USE_TDM 1
#endif
#endif
#ifndef USE_TDM
#define USE_TDM 0
#endif

#if USE_TDM
__device__ __forceinline__ void tdm_load_1d(unsigned lds_byte_addr,
                                            const void* gptr, int nelem2B)
{
    const unsigned long long ga = (unsigned long long)(size_t)gptr;
    v4u g0;
    g0[0] = 1u;                                              // count=1 (user D#)
    g0[1] = lds_byte_addr;                                   // lds_addr
    g0[2] = (unsigned)(ga & 0xFFFFFFFFull);                  // global_addr[31:0]
    g0[3] = (unsigned)((ga >> 32) & 0x01FFFFFFull)           // global_addr[56:32]
          | (2u << 30);                                      // type = 2 (image)
    v8i g1;
    g1[0] = (1 << 16);                                       // data_size=1 (2B)
    g1[1] = (int)((unsigned)(nelem2B & 0xFFFF) << 16);       // tensor_dim0 lo
    g1[2] = (int)(((unsigned)nelem2B >> 16) & 0xFFFF)        // tensor_dim0 hi
          | (1 << 16);                                       // tensor_dim1 = 1
    g1[3] = (int)((unsigned)(nelem2B & 0xFFFF) << 16);       // tile_dim0
    g1[4] = 0;                                               // tile_dim1/2 unused
    g1[5] = nelem2B;                                         // dim0_stride lo
    g1[6] = 0;
    g1[7] = 0;
    v4i gz = {0, 0, 0, 0};
#if __clang_major__ >= 23
    v8i gz8 = {0, 0, 0, 0, 0, 0, 0, 0};
    __builtin_amdgcn_tensor_load_to_lds(g0, g1, gz, gz, gz8, 0);
#else
    __builtin_amdgcn_tensor_load_to_lds(g0, g1, gz, gz, 0);
#endif
}
#endif

// ---------------------------------------------------------------------------
// Kernel 0: elementwise f32 -> bf16 (vectorized x4).
// ---------------------------------------------------------------------------
__global__ void __launch_bounds__(256)
cvt_f32_bf16_kernel(const float* __restrict__ src, __bf16* __restrict__ dst,
                    int n4)
{
    const int i = blockIdx.x * 256 + threadIdx.x;
    if (i < n4) {
        const float4 f = ((const float4*)src)[i];
        v4bf o;
        o[0] = (__bf16)f.x; o[1] = (__bf16)f.y;
        o[2] = (__bf16)f.z; o[3] = (__bf16)f.w;
        ((v4bf*)dst)[i] = o;
    }
}

// ---------------------------------------------------------------------------
// Kernel 1: Q/K/V/G projection, SEL as a compile-time parameter (no runtime
// selects in the epilogue). All 8 B fragments of the K-loop are loaded before
// the WMMA chain so loads stream underneath the XDL pipeline.
//   SEL 0: q (scaled by 1/sqrt(32)), SEL 1: k, SEL 2: v (stored TRANSPOSED
//   [s][h][c][k]), SEL 3: g (sigmoid via native rcp).
// ---------------------------------------------------------------------------
template<int SEL>
__global__ void __launch_bounds__(256)
proj_kernel(const __bf16* __restrict__ in, const __bf16* __restrict__ w,
            __bf16* __restrict__ outp)
{
    const int wave = threadIdx.x >> 5;
    const int lane = threadIdx.x & 31;
    const int l15  = lane & 15;
    const int lh   = lane >> 4;
    const int row0 = blockIdx.x * 16;

    // A fragments for the whole K range (shared by both column tiles)
    v16bf afr[8];
    #pragma unroll
    for (int kk = 0; kk < 8; ++kk)
        afr[kk] = load_a_frag(in + (size_t)(row0 + l15) * CQ + kk * 32, lh);

    for (int cpair = 0; cpair < 2; ++cpair) {
        const int ct = wave * 2 + cpair;
        const __bf16* wcol = w + (size_t)(ct * 16 + l15) * CQ + lh * 16;
        // burst-load all B fragments, then run the WMMA chain
        v16bf bfr[8];
        #pragma unroll
        for (int kk = 0; kk < 8; ++kk)
            bfr[kk] = load_b_frag(wcol + kk * 32);
        v8f acc = vzero8();
        #pragma unroll
        for (int kk = 0; kk < 8; ++kk)
            acc = wmma_bf16(afr[kk], bfr[kk], acc);

        #pragma unroll
        for (int r = 0; r < 8; ++r) {
            const int row = row0 + r + 8 * lh;
            const int col = ct * 16 + l15;
            float x = acc[r];
            if (SEL == 0) x *= 0.17677669529663687f;                    // 1/sqrt(32)
            if (SEL == 3) x = __builtin_amdgcn_rcpf(1.0f + __expf(-x)); // sigmoid
            const int s = row >> 8, qk = row & 255;
            const int h = col >> 5, cc = col & 31;
            if (SEL == 2)   // transposed V: [s][h][c][k]
                outp[((((size_t)s * HH) + h) * CH + cc) * 256 + qk] = (__bf16)x;
            else            // [s][h][row][c]
                outp[((((size_t)s * HH) + h) * 256 + qk) * CH + cc] = (__bf16)x;
        }
    }
}

// ---------------------------------------------------------------------------
// Kernel 2: attention per (s,h). grid=(128,8), block=256 (8 waves).
// TDM stages the 16KB K tile ([k][c]) and 16KB transposed V tile ([c][k])
// into LDS (reused by all 8 waves); all QK/PV fragments then come from LDS
// as contiguous 32B reads. Softmax in registers + half-wave shfl_xor.
// ---------------------------------------------------------------------------
__global__ void __launch_bounds__(256)
attn_kernel(const __bf16* __restrict__ qb, const __bf16* __restrict__ kb,
            const __bf16* __restrict__ vb, const __bf16* __restrict__ gb,
            const float* __restrict__ bias1, const float* __restrict__ bias2,
            __bf16* __restrict__ og)
{
    extern __shared__ char smem_raw[];
    __bf16* ldsK = (__bf16*)smem_raw;                  // [256][32] bf16, 16KB
    __bf16* ldsV = (__bf16*)(smem_raw + 16384);        // [32][256] bf16, 16KB
    __bf16* pall = (__bf16*)(smem_raw + 32768);        // 8 waves * [16][264]

    const int s    = blockIdx.x;
    const int h    = blockIdx.y;
    const int wave = threadIdx.x >> 5;
    const int lane = threadIdx.x & 31;
    const int l15  = lane & 15;
    const int lh   = lane >> 4;

    const size_t base = (((size_t)s * HH) + h) * 256 * CH;  // 8192-elem tile
    const __bf16* Qp = qb + base;
    const __bf16* Kp = kb + base;
    const __bf16* Vt = vb + base;   // transposed tile, same flat extent
    const __bf16* Gp = gb + base;
    __bf16* pw = pall + (size_t)wave * 16 * 264;

    // ---- stage K and V tiles into LDS ----
#if USE_TDM
    if (wave == 0) {
        const unsigned lbase = (unsigned)(size_t)smem_raw;
        tdm_load_1d(lbase,         Kp, 256 * CH);
        tdm_load_1d(lbase + 16384, Vt, 256 * CH);
        __builtin_amdgcn_s_wait_tensorcnt(0);
    }
    __syncthreads();
#else
    {
        const uint4* gk = (const uint4*)Kp;
        const uint4* gv = (const uint4*)Vt;
        uint4* lk = (uint4*)ldsK;
        uint4* lv = (uint4*)ldsV;
        for (int i = threadIdx.x; i < 1024; i += 256) { lk[i] = gk[i]; lv[i] = gv[i]; }
        __syncthreads();
    }
#endif

    const float* b1 = bias1 + (size_t)s * NKK;
    const float* b2 = bias2 + (size_t)h * NQ * NKK;

    for (int qt2 = 0; qt2 < 2; ++qt2) {
        const int q0 = wave * 32 + qt2 * 16;

        // ---- logits: one WMMA per 16x16 tile; 16 independent accumulators ----
        v16bf a = load_a_frag(Qp + (size_t)(q0 + l15) * CH, lh);
        v8f acc[16];
        #pragma unroll
        for (int nt = 0; nt < 16; ++nt) {
            v16bf b = load_b_frag(ldsK + (size_t)(nt * 16 + l15) * CH + lh * 16);
            acc[nt] = wmma_bf16(a, b, vzero8());
        }
        // ---- biases ----
        #pragma unroll
        for (int nt = 0; nt < 16; ++nt) {
            const int kr = nt * 16 + l15;
            const float bb1 = b1[kr];
            #pragma unroll
            for (int r = 0; r < 8; ++r) {
                const int qr = q0 + r + 8 * lh;
                acc[nt][r] += bb1 + b2[(size_t)qr * NKK + kr];
            }
        }
        // ---- softmax per row (r, half); shfl_xor stays within half ----
        #pragma unroll
        for (int r = 0; r < 8; ++r) {
            float m = -1e30f;
            #pragma unroll
            for (int nt = 0; nt < 16; ++nt) m = fmaxf(m, acc[nt][r]);
            for (int off = 8; off >= 1; off >>= 1)
                m = fmaxf(m, __shfl_xor(m, off, 32));
            float sum = 0.f;
            #pragma unroll
            for (int nt = 0; nt < 16; ++nt) {
                const float p = __expf(acc[nt][r] - m);
                acc[nt][r] = p;
                sum += p;
            }
            for (int off = 8; off >= 1; off >>= 1)
                sum += __shfl_xor(sum, off, 32);
            const float inv = __builtin_amdgcn_rcpf(sum);
            #pragma unroll
            for (int nt = 0; nt < 16; ++nt) acc[nt][r] *= inv;
        }
        // ---- stage P (bf16) in per-wave LDS slab [16][264] ----
        #pragma unroll
        for (int nt = 0; nt < 16; ++nt)
            #pragma unroll
            for (int r = 0; r < 8; ++r)
                pw[(r + 8 * lh) * 264 + nt * 16 + l15] = (__bf16)acc[nt][r];
        // per-wave private slab; LDS ops are in-order within a wave

        // ---- O = P(16x256) * V(256x32), V B-fragments contiguous in ldsV ----
        v8f oacc0 = vzero8(), oacc1 = vzero8();
        #pragma unroll
        for (int kt = 0; kt < 8; ++kt) {
            v16bf pa = load_a_frag(pw + (size_t)l15 * 264 + kt * 32, lh);
            v16bf vf0 = load_b_frag(ldsV + (size_t)(l15) * 256 + kt * 32 + lh * 16);
            v16bf vf1 = load_b_frag(ldsV + (size_t)(16 + l15) * 256 + kt * 32 + lh * 16);
            oacc0 = wmma_bf16(pa, vf0, oacc0);
            oacc1 = wmma_bf16(pa, vf1, oacc1);
        }
        // ---- gate + store bf16 gated output [s][h][q][c] ----
        #pragma unroll
        for (int ct = 0; ct < 2; ++ct)
            #pragma unroll
            for (int r = 0; r < 8; ++r) {
                const int qr = q0 + r + 8 * lh;
                const int c  = ct * 16 + l15;
                const float gv = (float)Gp[(size_t)qr * CH + c];
                const float ov = (ct == 0) ? oacc0[r] : oacc1[r];
                og[base + (size_t)qr * CH + c] = (__bf16)(ov * gv);
            }
    }
}

// ---------------------------------------------------------------------------
// Kernel 3: out = gated_o_flat[32768,256] @ wo^T (f32 out). A operand gathered
// from og's [s][h][q][c] layout as two contiguous 16B runs per lane; all B
// fragments burst-loaded ahead of the WMMA chain.
// ---------------------------------------------------------------------------
__global__ void __launch_bounds__(256)
outproj_kernel(const __bf16* __restrict__ og, const __bf16* __restrict__ wo,
               float* __restrict__ out)
{
    const int wave = threadIdx.x >> 5;
    const int lane = threadIdx.x & 31;
    const int l15  = lane & 15;
    const int lh   = lane >> 4;
    const int row0 = blockIdx.x * 16;

    const int n = row0 + l15;
    const int s = n >> 8, qq = n & 255;

    // A fragments for the whole K range
    v16bf afr[8];
    #pragma unroll
    for (int kk = 0; kk < 8; ++kk) {
        const int k0 = kk * 32 + lh * 16;
        const int k1 = kk * 32 + lh * 16 + 8;
        v8bf lo = *(const v8bf*)(og +
            ((((size_t)s * HH) + (k0 >> 5)) * 256 + qq) * CH + (k0 & 31));
        v8bf hi = *(const v8bf*)(og +
            ((((size_t)s * HH) + (k1 >> 5)) * 256 + qq) * CH + (k1 & 31));
        #pragma unroll
        for (int j = 0; j < 8; ++j) { afr[kk][j] = lo[j]; afr[kk][8 + j] = hi[j]; }
    }

    for (int cpair = 0; cpair < 2; ++cpair) {
        const int ct = wave * 2 + cpair;
        const __bf16* wcol = wo + (size_t)(ct * 16 + l15) * CQ + lh * 16;
        v16bf bfr[8];
        #pragma unroll
        for (int kk = 0; kk < 8; ++kk)
            bfr[kk] = load_b_frag(wcol + kk * 32);
        v8f acc = vzero8();
        #pragma unroll
        for (int kk = 0; kk < 8; ++kk)
            acc = wmma_bf16(afr[kk], bfr[kk], acc);

        #pragma unroll
        for (int r = 0; r < 8; ++r) {
            const int nn  = row0 + r + 8 * lh;
            const int col = ct * 16 + l15;
            out[(size_t)nn * CQ + col] = acc[r];
        }
    }
}

// ---------------------------------------------------------------------------
extern "C" void kernel_launch(void* const* d_in, const int* in_sizes, int n_in,
                              void* d_out, int out_size, void* d_ws, size_t ws_size,
                              hipStream_t stream) {
    (void)in_sizes; (void)n_in; (void)out_size; (void)ws_size;
    const float* q_x   = (const float*)d_in[0];
    const float* kv_x  = (const float*)d_in[1];
    const float* bias1 = (const float*)d_in[2];
    const float* bias2 = (const float*)d_in[3];
    const float* wq    = (const float*)d_in[4];
    const float* wk    = (const float*)d_in[5];
    const float* wv    = (const float*)d_in[6];
    const float* wg    = (const float*)d_in[7];
    const float* wo    = (const float*)d_in[8];
    float* out = (float*)d_out;

    // ws layout (bf16 elems): q,k,v,g,og tiles + bf16 copies of inputs/weights.
    // ~118 MB total -> whole working set is L2-resident (192 MB).
    const size_t PER = (size_t)SS * HH * 256 * CH;   // 8,388,608
    const size_t WSZ = (size_t)CQ * CQ;              // 65,536
    __bf16* ws   = (__bf16*)d_ws;
    __bf16* qb   = ws + 0 * PER;
    __bf16* kb   = ws + 1 * PER;
    __bf16* vb   = ws + 2 * PER;   // transposed per-(s,h): [c][k]
    __bf16* gb   = ws + 3 * PER;
    __bf16* og   = ws + 4 * PER;
    __bf16* qxb  = ws + 5 * PER;
    __bf16* kvxb = ws + 6 * PER;
    __bf16* wqb  = ws + 7 * PER;
    __bf16* wkb  = wqb + WSZ;
    __bf16* wvb  = wkb + WSZ;
    __bf16* wgb  = wvb + WSZ;
    __bf16* wob  = wgb + WSZ;

    // 0) one-shot bf16 conversion of activations + weights
    {
        const int n4a = (int)(PER / 4), n4w = (int)(WSZ / 4);
        cvt_f32_bf16_kernel<<<(n4a + 255) / 256, 256, 0, stream>>>(q_x,  qxb,  n4a);
        cvt_f32_bf16_kernel<<<(n4a + 255) / 256, 256, 0, stream>>>(kv_x, kvxb, n4a);
        cvt_f32_bf16_kernel<<<(n4w + 255) / 256, 256, 0, stream>>>(wq, wqb, n4w);
        cvt_f32_bf16_kernel<<<(n4w + 255) / 256, 256, 0, stream>>>(wk, wkb, n4w);
        cvt_f32_bf16_kernel<<<(n4w + 255) / 256, 256, 0, stream>>>(wv, wvb, n4w);
        cvt_f32_bf16_kernel<<<(n4w + 255) / 256, 256, 0, stream>>>(wg, wgb, n4w);
        cvt_f32_bf16_kernel<<<(n4w + 255) / 256, 256, 0, stream>>>(wo, wob, n4w);
    }
    // 1) projections (compile-time SEL, no runtime selects)
    {
        dim3 grid((SS * NQ) / 16);
        proj_kernel<0><<<grid, dim3(256), 0, stream>>>(qxb,  wqb, qb);
        proj_kernel<1><<<grid, dim3(256), 0, stream>>>(kvxb, wkb, kb);
        proj_kernel<2><<<grid, dim3(256), 0, stream>>>(kvxb, wvb, vb);
        proj_kernel<3><<<grid, dim3(256), 0, stream>>>(qxb,  wgb, gb);
    }
    // 2) attention + gating per (s,h); LDS = K tile + V tile + P slabs
    {
        dim3 grid(SS, HH);
        const size_t smem = 16384 + 16384 + (size_t)8 * 16 * 264 * sizeof(__bf16);
        attn_kernel<<<grid, dim3(256), smem, stream>>>(
            qb, kb, vb, gb, bias1, bias2, og);
    }
    // 3) output projection
    {
        dim3 grid((SS * NQ) / 16);
        outproj_kernel<<<grid, dim3(256), 0, stream>>>(og, wob, out);
    }
}
